// _MambaSSM_82471962018364
// MI455X (gfx1250) — matrix operand order repaired and checked by reading, not compiled, not run on hardware
//
#include <hip/hip_runtime.h>
#include <hip/hip_bf16.h>

// ---------------------------------------------------------------------------
// Mamba SSM for MI455X (gfx1250, wave32, WMMA)
//   Phase 1: x_proj GEMM (8192x2080x2048) via v_wmma_f32_16x16x32_f16.
//            Software-pipelined, double-buffered LDS (TK=64 per stage):
//            global_load_b128 for tile i+1 issue BEFORE the barrier, WMMAs
//            for tile i run while loads are in flight, then cvt+ds_store
//            into the other buffer. One barrier per stage.
//   Phase 2: chunked linear-recurrence scan (16 chunks of 128):
//            part1 = per-chunk transfer (P, h_loc); part2 = fold + apply.
// ---------------------------------------------------------------------------

typedef __attribute__((ext_vector_type(16))) _Float16 v16h;
typedef __attribute__((ext_vector_type(8)))  float    v8f;

#define D_INNER 2048
#define D_STATE 16
#define E_DIM   (D_INNER + 2 * D_STATE)   // 2080
#define BATCH   4
#define SEQ     2048
#define M_DIM   (BATCH * SEQ)             // 8192

// GEMM tiling
#define TM 128          // block M tile
#define TN 64           // block N tile
#define TK 64           // LDS stage depth (= 2 WMMA K-steps)
#define LDS_PAD 72      // LDS row stride in halves (64 data + 8 pad)
#define NSTAGE  (D_INNER / TK)            // 32 stages

// Scan chunking
#define CHUNKS 16
#define LC     (SEQ / CHUNKS)             // 128 steps per chunk

__device__ __forceinline__ float softplus_f(float x) {
    return (x > 20.0f) ? x : __logf(1.0f + __expf(x));
}

// ---------------------------------------------------------------------------
// x_proj GEMM (double-buffered pipeline)
// ---------------------------------------------------------------------------

// Issue next tile's global loads into registers and advance uniform bases.
#define GEMM_STAGE_LOAD()                                                   \
    {                                                                       \
        _Pragma("unroll")                                                   \
        for (int i = 0; i < 8; ++i)                                         \
            aReg[i] = *(const float4*)(Xb + aOffB[i]);                      \
        _Pragma("unroll")                                                   \
        for (int i = 0; i < 4; ++i)                                         \
            bReg[i] = *(const float4*)(Wb + bOffB[i]);                      \
        Xb += TK * 4;                                                       \
        Wb += TK * 4;                                                       \
        __builtin_prefetch(Xb + aOffB[0], 0, 1);  /* global_prefetch_b8 */  \
    }

// Convert staged registers to f16 and store into LDS buffer P (compile-time
// P folds into ds_store immediate offsets).
#define GEMM_STAGE_STORE(P)                                                 \
    {                                                                       \
        _Pragma("unroll")                                                   \
        for (int i = 0; i < 8; ++i) {                                       \
            _Float16* dp = aDst[i] + (P) * (TM * LDS_PAD);                  \
            dp[0] = (_Float16)aReg[i].x; dp[1] = (_Float16)aReg[i].y;       \
            dp[2] = (_Float16)aReg[i].z; dp[3] = (_Float16)aReg[i].w;       \
        }                                                                   \
        _Pragma("unroll")                                                   \
        for (int i = 0; i < 4; ++i) {                                       \
            _Float16* dp = bDst[i] + (P) * (TN * LDS_PAD);                  \
            dp[0] = (_Float16)bReg[i].x; dp[1] = (_Float16)bReg[i].y;       \
            dp[2] = (_Float16)bReg[i].z; dp[3] = (_Float16)bReg[i].w;       \
        }                                                                   \
    }

// 2 WMMA K-steps (8 wmma) from LDS buffer P.
#define GEMM_COMPUTE(P)                                                     \
    {                                                                       \
        _Pragma("unroll")                                                   \
        for (int ks = 0; ks < TK; ks += 32) {                               \
            v16h afrag[2];                                                  \
            _Pragma("unroll")                                               \
            for (int tm = 0; tm < 2; ++tm) {                                \
                const _Float16* arow = As[P][wm * 32 + tm * 16 + mrow];     \
                _Pragma("unroll")                                           \
                for (int j = 0; j < 8; ++j) {                               \
                    int kb = ks + ((j >> 2) << 4) + (halfw << 3) + ((j & 3) << 1); \
                    afrag[tm][2 * j]     = arow[kb];                        \
                    afrag[tm][2 * j + 1] = arow[kb + 1];                    \
                }                                                           \
            }                                                               \
            v16h bfrag[2];                                                  \
            _Pragma("unroll")                                               \
            for (int tn = 0; tn < 2; ++tn) {                                \
                const _Float16* brow = Bs[P][wn * 32 + tn * 16 + mrow];     \
                _Pragma("unroll")                                           \
                for (int j = 0; j < 8; ++j) {                               \
                    int kb = ks + ((j >> 2) << 4) + (halfw << 3) + ((j & 3) << 1); \
                    bfrag[tn][2 * j]     = brow[kb];                        \
                    bfrag[tn][2 * j + 1] = brow[kb + 1];                    \
                }                                                           \
            }                                                               \
            _Pragma("unroll")                                               \
            for (int tm = 0; tm < 2; ++tm) {                                \
                _Pragma("unroll")                                           \
                for (int tn = 0; tn < 2; ++tn) {                            \
                    acc[tm][tn] = __builtin_amdgcn_wmma_f32_16x16x32_f16(   \
                        false, afrag[tm], false, bfrag[tn],                 \
                        (short)0, acc[tm][tn], false, false);               \
                }                                                           \
            }                                                               \
        }                                                                   \
    }

__global__ __launch_bounds__(256)
void xproj_gemm_kernel(const float* __restrict__ X,   // [M_DIM, D_INNER]
                       const float* __restrict__ W,   // [E_DIM, D_INNER]
                       float* __restrict__ XP)        // [M_DIM, E_DIM]
{
    __shared__ _Float16 As[2][TM][LDS_PAD];   // 36,864 B
    __shared__ _Float16 Bs[2][TN][LDS_PAD];   // 18,432 B

    const int tid   = threadIdx.x;
    const int lane  = tid & 31;
    const int wave  = tid >> 5;
    const int wm    = wave & 3;       // wave M offset: wm*32
    const int wn    = wave >> 2;      // wave N offset: wn*32
    const int halfw = lane >> 4;
    const int mrow  = lane & 15;

    const int n0 = blockIdx.x * TN;
    const int m0 = blockIdx.y * TM;

    // Uniform byte base pointers (advance = scalar adds) + loop-invariant
    // 32-bit per-thread byte offsets.
    const char* Xb = (const char*)X;
    const char* Wb = (const char*)W;

    unsigned  aOffB[8];
    _Float16* aDst[8];
    #pragma unroll
    for (int i = 0; i < 8; ++i) {                 // A tile: 128x64 = 2048 float4
        int li = tid + 256 * i;
        int r = li >> 4, c4 = li & 15;
        aOffB[i] = (unsigned)(((m0 + r) * (D_INNER / 4) + c4) * 16);
        aDst[i]  = &As[0][r][c4 * 4];
    }
    unsigned  bOffB[4];
    _Float16* bDst[4];
    #pragma unroll
    for (int i = 0; i < 4; ++i) {                 // B tile: 64x64 = 1024 float4
        int li = tid + 256 * i;
        int r = li >> 4, c4 = li & 15;
        int e = n0 + r;
        if (e > E_DIM - 1) e = E_DIM - 1;         // clamp: garbage cols never stored
        bOffB[i] = (unsigned)((e * (D_INNER / 4) + c4) * 16);
        bDst[i]  = &Bs[0][r][c4 * 4];
    }

    v8f acc[2][2];
    {
        v8f z = {};
        acc[0][0] = z; acc[0][1] = z; acc[1][0] = z; acc[1][1] = z;
    }

    float4 aReg[8], bReg[4];

    // Pipeline: prologue stages tile 0 into buffer 0.
    GEMM_STAGE_LOAD();                 // tile 0
    GEMM_STAGE_STORE(0);

    // Steady state: 2 stages per iteration (ping-pong buffers).
    for (int it = 0; it < NSTAGE / 2 - 1; ++it) {   // 15 iterations
        GEMM_STAGE_LOAD();             // tile 2it+1
        __syncthreads();
        GEMM_COMPUTE(0);
        GEMM_STAGE_STORE(1);
        GEMM_STAGE_LOAD();             // tile 2it+2
        __syncthreads();
        GEMM_COMPUTE(1);
        GEMM_STAGE_STORE(0);
    }
    // Epilogue: last load (tile 31), compute both remaining buffers.
    GEMM_STAGE_LOAD();                 // tile 31
    __syncthreads();
    GEMM_COMPUTE(0);
    GEMM_STAGE_STORE(1);
    __syncthreads();
    GEMM_COMPUTE(1);

    #pragma unroll
    for (int tm = 0; tm < 2; ++tm) {
        #pragma unroll
        for (int tn = 0; tn < 2; ++tn) {
            int colb = n0 + wn * 32 + tn * 16;
            if (colb < E_DIM) {
                #pragma unroll
                for (int j = 0; j < 8; ++j) {
                    int row = m0 + wm * 32 + tm * 16 + halfw * 8 + j;
                    XP[(size_t)row * E_DIM + colb + mrow] = acc[tm][tn][j];
                }
            }
        }
    }
}

// ---------------------------------------------------------------------------
// Scan phase 1: per-chunk transfer function.
// Thread = (b, d, chunk). Computes P[n] = prod(abar), h_loc[n] (h_in = 0).
// Summary layout: P/H[((b*CHUNKS + c)*D_STATE + n)*D_INNER + d]
// ---------------------------------------------------------------------------
__global__ __launch_bounds__(256)
void scan_part1_kernel(const float* __restrict__ X,
                       const float* __restrict__ Alog,
                       const float* __restrict__ XP,
                       float* __restrict__ Pbuf,
                       float* __restrict__ Hbuf)
{
    const int d = blockIdx.x * blockDim.x + threadIdx.x;
    const int b = blockIdx.y;
    const int c = blockIdx.z;

    float A[D_STATE], P[D_STATE], h[D_STATE];
    #pragma unroll
    for (int n = 0; n < D_STATE; ++n) {
        A[n] = -__expf(Alog[d * D_STATE + n]);
        P[n] = 1.0f;
        h[n] = 0.0f;
    }

    const int l0 = c * LC;
    for (int l = l0; l < l0 + LC; ++l) {
        const size_t row = (size_t)b * SEQ + l;
        const float* xprow = XP + row * E_DIM;
        float dt = softplus_f(xprow[d]);
        float xv = X[row * D_INNER + d];

        const float4* B4 = (const float4*)(xprow + D_INNER);
        float Bv[D_STATE];
        #pragma unroll
        for (int q = 0; q < 4; ++q) {
            float4 bb = B4[q];
            Bv[4 * q + 0] = bb.x; Bv[4 * q + 1] = bb.y;
            Bv[4 * q + 2] = bb.z; Bv[4 * q + 3] = bb.w;
        }
        #pragma unroll
        for (int n = 0; n < D_STATE; ++n) {
            float abar = __expf(dt * A[n]);
            P[n] *= abar;
            h[n] = fmaf(abar, h[n], (dt * Bv[n]) * xv);
        }
    }

    const size_t sbase = ((size_t)(b * CHUNKS + c) * D_STATE) * D_INNER + d;
    #pragma unroll
    for (int n = 0; n < D_STATE; ++n) {
        Pbuf[sbase + (size_t)n * D_INNER] = P[n];
        Hbuf[sbase + (size_t)n * D_INNER] = h[n];
    }
}

// ---------------------------------------------------------------------------
// Scan phase 2: fold preceding chunk summaries -> h_in, then re-run the
// recurrence for this chunk with the correct initial state and emit y.
// ---------------------------------------------------------------------------
__global__ __launch_bounds__(256)
void scan_part2_kernel(const float* __restrict__ X,
                       const float* __restrict__ Alog,
                       const float* __restrict__ Dp,
                       const float* __restrict__ XP,
                       const float* __restrict__ Pbuf,
                       const float* __restrict__ Hbuf,
                       float* __restrict__ Y)
{
    const int d = blockIdx.x * blockDim.x + threadIdx.x;
    const int b = blockIdx.y;
    const int c = blockIdx.z;

    float A[D_STATE], h[D_STATE];
    #pragma unroll
    for (int n = 0; n < D_STATE; ++n) {
        A[n] = -__expf(Alog[d * D_STATE + n]);
        h[n] = 0.0f;
    }
    const float Dv = Dp[d];

    // fold summaries of chunks 0..c-1 (uniform trip count per block)
    for (int cc = 0; cc < c; ++cc) {
        const size_t sbase = ((size_t)(b * CHUNKS + cc) * D_STATE) * D_INNER + d;
        #pragma unroll
        for (int n = 0; n < D_STATE; ++n) {
            float Pn = Pbuf[sbase + (size_t)n * D_INNER];
            float Hn = Hbuf[sbase + (size_t)n * D_INNER];
            h[n] = fmaf(Pn, h[n], Hn);
        }
    }

    const int l0 = c * LC;
    for (int l = l0; l < l0 + LC; ++l) {
        const size_t row = (size_t)b * SEQ + l;
        const float* xprow = XP + row * E_DIM;
        float dt = softplus_f(xprow[d]);
        float xv = X[row * D_INNER + d];

        const float4* B4 = (const float4*)(xprow + D_INNER);
        const float4* C4 = (const float4*)(xprow + D_INNER + D_STATE);
        float Bv[D_STATE], Cv[D_STATE];
        #pragma unroll
        for (int q = 0; q < 4; ++q) {
            float4 bb = B4[q], cc4 = C4[q];
            Bv[4 * q + 0] = bb.x;  Bv[4 * q + 1] = bb.y;
            Bv[4 * q + 2] = bb.z;  Bv[4 * q + 3] = bb.w;
            Cv[4 * q + 0] = cc4.x; Cv[4 * q + 1] = cc4.y;
            Cv[4 * q + 2] = cc4.z; Cv[4 * q + 3] = cc4.w;
        }
        float y = 0.0f;
        #pragma unroll
        for (int n = 0; n < D_STATE; ++n) {
            float abar = __expf(dt * A[n]);
            h[n] = fmaf(abar, h[n], (dt * Bv[n]) * xv);
            y = fmaf(h[n], Cv[n], y);
        }
        Y[row * D_INNER + d] = fmaf(Dv, xv, y);
    }
}

// ---------------------------------------------------------------------------
extern "C" void kernel_launch(void* const* d_in, const int* in_sizes, int n_in,
                              void* d_out, int out_size, void* d_ws, size_t ws_size,
                              hipStream_t stream) {
    const float* x      = (const float*)d_in[0];  // [4, 2048, 2048]
    const float* A_log  = (const float*)d_in[1];  // [2048, 16]
    const float* Dvec   = (const float*)d_in[2];  // [2048]
    const float* Wxproj = (const float*)d_in[3];  // [2080, 2048]
    float* out = (float*)d_out;                   // [4, 2048, 2048]

    // workspace layout (floats):
    //   xp:   M_DIM * E_DIM                = 17,039,360  (68.2 MB)
    //   Pbuf: BATCH*CHUNKS*D_STATE*D_INNER =  2,097,152  ( 8.4 MB)
    //   Hbuf: same                                        ( 8.4 MB)
    float* xp   = (float*)d_ws;
    float* Pbuf = xp + (size_t)M_DIM * E_DIM;
    float* Hbuf = Pbuf + (size_t)BATCH * CHUNKS * D_STATE * D_INNER;

    dim3 ggrid((E_DIM + TN - 1) / TN, M_DIM / TM);   // (33, 64)
    xproj_gemm_kernel<<<ggrid, 256, 0, stream>>>(x, Wxproj, xp);

    dim3 sgrid(D_INNER / 256, BATCH, CHUNKS);        // (8, 4, 16)
    scan_part1_kernel<<<sgrid, 256, 0, stream>>>(x, A_log, xp, Pbuf, Hbuf);
    scan_part2_kernel<<<sgrid, 256, 0, stream>>>(x, A_log, Dvec, xp, Pbuf, Hbuf, out);
}
